// GraphSageEncoder_41523743817923
// MI455X (gfx1250) — matrix operand order repaired and checked by reading, not compiled, not run on hardware
//
#include <hip/hip_runtime.h>

#define DIN 256
#define HD  128

typedef __attribute__((ext_vector_type(16))) __bf16          v16bf;
typedef __attribute__((ext_vector_type(8)))  float           v8f;
typedef __attribute__((ext_vector_type(8)))  unsigned short  us8;

union AFrag { v16bf v; us8 h[2]; unsigned short u[16]; };

__device__ __forceinline__ unsigned short f2bf(float f) {
    unsigned u = __builtin_bit_cast(unsigned, f);
    u += 0x7FFFu + ((u >> 16) & 1u);          // round-to-nearest-even
    return (unsigned short)(u >> 16);
}

// ---------------------------------------------------------------------------
// Convert + transpose weights fp32 -> bf16.  Layout in `o` (ushort units):
//   [0)      W_in^T  [128 x 256]
//   [32768)  Wl1^T   [128 x 128]
//   [49152)  Wr1^T   [128 x 128]
//   [65536)  Wl2^T   [128 x 128]
//   [81920)  Wr2^T   [128 x 128]
// ---------------------------------------------------------------------------
__global__ __launch_bounds__(256) void cvt_kernel(
    const float* __restrict__ Win, const float* __restrict__ Wl1,
    const float* __restrict__ Wr1, const float* __restrict__ Wl2,
    const float* __restrict__ Wr2, unsigned short* __restrict__ o)
{
    int i = blockIdx.x * 256 + threadIdx.x;
    if (i < 32768) {                       // W_in [256 x 128] row-major
        int k = i >> 7, n = i & 127;
        o[n * 256 + k] = f2bf(Win[i]);
    } else if (i < 98304) {
        int j = i - 32768;
        int m = j >> 14;                   // which 128x128 matrix
        int r = j & 16383;
        int k = r >> 7, n = r & 127;
        const float* src = (m == 0) ? Wl1 : (m == 1) ? Wr1 : (m == 2) ? Wl2 : Wr2;
        o[32768 + m * 16384 + n * 128 + k] = f2bf(src[r]);
    }
}

// ---------------------------------------------------------------------------
// h0 = x @ W_in + b_in + kind_embed[kind] + (kind!=0)*ext_seed
// block = 256 thr = 8 waves; block row-tile = 16 rows; wave w -> cols [16w,16w+16)
// ---------------------------------------------------------------------------
__global__ __launch_bounds__(256) void encode_kernel(
    const float* __restrict__ x, const float* __restrict__ b_in,
    const int* __restrict__ kind, const float* __restrict__ kind_embed,
    const float* __restrict__ ext_seed, const unsigned short* __restrict__ Wt,
    float* __restrict__ h0, int nnodes)
{
    __shared__ __attribute__((aligned(16))) unsigned short sA[16][264]; // 256+8 pad
    const int row0 = blockIdx.x * 16;
    const int tid  = threadIdx.x;

    for (int i = tid; i < 16 * 256; i += 256) {
        int r = i >> 8, c = i & 255;
        int rr = min(row0 + r, nnodes - 1);
        sA[r][c] = f2bf(x[(size_t)rr * DIN + c]);
    }
    __syncthreads();

    const int wave = tid >> 5;
    const int lane = tid & 31;
    const int hf   = lane >> 4;          // half-wave select
    const int l16  = lane & 15;
    const int col0 = wave * 16;
    const int n    = col0 + l16;

    v8f acc = {};
    #pragma unroll
    for (int kk = 0; kk < 8; ++kk) {     // K = 256, step 32
        const int k0 = kk * 32;
        AFrag a, b;
        const unsigned short* ap = &sA[l16][k0 + hf * 8];
        a.h[0] = *(const us8*)(ap);
        a.h[1] = *(const us8*)(ap + 16);
        b.h[0] = *(const us8*)(Wt + (size_t)n * 256 + k0 + hf * 8);
        b.h[1] = *(const us8*)(Wt + (size_t)n * 256 + k0 + hf * 8 + 16);
        acc = __builtin_amdgcn_wmma_f32_16x16x32_bf16(
                  false, a.v, false, b.v, (short)0, acc, false, false);
    }

    #pragma unroll
    for (int r = 0; r < 8; ++r) {
        int m = row0 + hf * 8 + r;
        if (m < nnodes) {
            int kd = kind[m];
            float v = acc[r] + b_in[n] + kind_embed[kd * HD + n] +
                      (kd != 0 ? ext_seed[n] : 0.f);
            h0[(size_t)m * HD + n] = v;
        }
    }
}

// ---------------------------------------------------------------------------
__global__ __launch_bounds__(256) void zero_kernel(
    float* __restrict__ agg, float* __restrict__ cnt, int nnodes)
{
    int i = blockIdx.x * 256 + threadIdx.x;
    if (i < nnodes * HD) agg[i] = 0.f;
    if (i < nnodes)      cnt[i] = 0.f;
}

// ---------------------------------------------------------------------------
// One wave per edge: gather h[src] (float4/lane), scatter-add to agg[dst].
// ---------------------------------------------------------------------------
__global__ __launch_bounds__(256) void scatter_kernel(
    const float* __restrict__ h, const int* __restrict__ src,
    const int* __restrict__ dst, float* __restrict__ agg,
    float* __restrict__ cnt, int nedges)
{
    int e = blockIdx.x * 8 + (threadIdx.x >> 5);
    if (e >= nedges) return;
    int lane = threadIdx.x & 31;
    int s = src[e], d = dst[e];
    const float* hp = h + (size_t)s * HD + lane * 4;
    float* ap = agg + (size_t)d * HD + lane * 4;
    __builtin_prefetch(ap, 1, 3);
    float4 v = *(const float4*)hp;
    atomicAdd(ap + 0, v.x);
    atomicAdd(ap + 1, v.y);
    atomicAdd(ap + 2, v.z);
    atomicAdd(ap + 3, v.w);
    if (lane == 0) atomicAdd(&cnt[d], 1.f);
}

// ---------------------------------------------------------------------------
// out = (agg / max(cnt,1)) @ Wl + hprev @ Wr + bias     (dual WMMA chain)
// ---------------------------------------------------------------------------
__global__ __launch_bounds__(256) void sage_kernel(
    const float* __restrict__ agg, const float* __restrict__ cnt,
    const float* __restrict__ hprev,
    const unsigned short* __restrict__ Wlt, const unsigned short* __restrict__ Wrt,
    const float* __restrict__ bias, float* __restrict__ out, int nnodes)
{
    __shared__ __attribute__((aligned(16))) unsigned short sAgg[16][136];
    __shared__ __attribute__((aligned(16))) unsigned short sH[16][136];
    const int row0 = blockIdx.x * 16;
    const int tid  = threadIdx.x;

    for (int i = tid; i < 16 * HD; i += 256) {
        int r = i >> 7, c = i & 127;
        int rr = min(row0 + r, nnodes - 1);
        float cn = cnt[rr];
        cn = cn > 1.f ? cn : 1.f;
        sAgg[r][c] = f2bf(agg[(size_t)rr * HD + c] / cn);
        sH[r][c]   = f2bf(hprev[(size_t)rr * HD + c]);
    }
    __syncthreads();

    const int wave = tid >> 5;
    const int lane = tid & 31;
    const int hf   = lane >> 4;
    const int l16  = lane & 15;
    const int n    = wave * 16 + l16;

    v8f acc = {};
    #pragma unroll
    for (int kk = 0; kk < 4; ++kk) {     // K = 128, step 32
        const int k0 = kk * 32;
        AFrag a, b;
        const unsigned short* ap = &sAgg[l16][k0 + hf * 8];
        a.h[0] = *(const us8*)(ap);
        a.h[1] = *(const us8*)(ap + 16);
        b.h[0] = *(const us8*)(Wlt + (size_t)n * HD + k0 + hf * 8);
        b.h[1] = *(const us8*)(Wlt + (size_t)n * HD + k0 + hf * 8 + 16);
        acc = __builtin_amdgcn_wmma_f32_16x16x32_bf16(
                  false, a.v, false, b.v, (short)0, acc, false, false);

        const unsigned short* hp = &sH[l16][k0 + hf * 8];
        a.h[0] = *(const us8*)(hp);
        a.h[1] = *(const us8*)(hp + 16);
        b.h[0] = *(const us8*)(Wrt + (size_t)n * HD + k0 + hf * 8);
        b.h[1] = *(const us8*)(Wrt + (size_t)n * HD + k0 + hf * 8 + 16);
        acc = __builtin_amdgcn_wmma_f32_16x16x32_bf16(
                  false, a.v, false, b.v, (short)0, acc, false, false);
    }

    #pragma unroll
    for (int r = 0; r < 8; ++r) {
        int m = row0 + hf * 8 + r;
        if (m < nnodes) out[(size_t)m * HD + n] = acc[r] + bias[n];
    }
}

// ---------------------------------------------------------------------------
// In-place ReLU(LayerNorm(row)).  One wave32 per row, 4 floats per lane.
// ---------------------------------------------------------------------------
__global__ __launch_bounds__(256) void ln_relu_kernel(
    float* __restrict__ t, const float* __restrict__ g,
    const float* __restrict__ b, int nnodes)
{
    int row = blockIdx.x * 8 + (threadIdx.x >> 5);
    if (row >= nnodes) return;
    int lane = threadIdx.x & 31;
    int c = lane * 4;
    float4 v = *(const float4*)(t + (size_t)row * HD + c);

    float s = v.x + v.y + v.z + v.w;
    #pragma unroll
    for (int o = 16; o; o >>= 1) s += __shfl_xor(s, o, 32);
    float mu = s * (1.f / 128.f);

    float dx = v.x - mu, dy = v.y - mu, dz = v.z - mu, dw = v.w - mu;
    float q = dx * dx + dy * dy + dz * dz + dw * dw;
    #pragma unroll
    for (int o = 16; o; o >>= 1) q += __shfl_xor(q, o, 32);
    float inv = rsqrtf(q * (1.f / 128.f) + 1e-5f);

    float4 r;
    r.x = fmaxf(dx * inv * g[c + 0] + b[c + 0], 0.f);
    r.y = fmaxf(dy * inv * g[c + 1] + b[c + 1], 0.f);
    r.z = fmaxf(dz * inv * g[c + 2] + b[c + 2], 0.f);
    r.w = fmaxf(dw * inv * g[c + 3] + b[c + 3], 0.f);
    *(float4*)(t + (size_t)row * HD + c) = r;
}

// ---------------------------------------------------------------------------
extern "C" void kernel_launch(void* const* d_in, const int* in_sizes, int n_in,
                              void* d_out, int out_size, void* d_ws, size_t ws_size,
                              hipStream_t stream)
{
    const float* x          = (const float*)d_in[0];
    const int*   edge_index = (const int*)  d_in[1];
    const int*   node_kind  = (const int*)  d_in[2];
    const float* W_in       = (const float*)d_in[3];
    const float* b_in       = (const float*)d_in[4];
    const float* kind_embed = (const float*)d_in[5];
    const float* ext_seed   = (const float*)d_in[6];
    const float* Wl1        = (const float*)d_in[7];
    const float* bl1        = (const float*)d_in[8];
    const float* Wr1        = (const float*)d_in[9];
    const float* gamma      = (const float*)d_in[10];
    const float* beta       = (const float*)d_in[11];
    const float* Wl2        = (const float*)d_in[12];
    const float* bl2        = (const float*)d_in[13];
    const float* Wr2        = (const float*)d_in[14];
    float* out = (float*)d_out;

    const int N = in_sizes[0] / DIN;          // 50000
    const int E = in_sizes[1] / 2;            // 625000
    const int* esrc = edge_index;
    const int* edst = edge_index + E;

    // workspace layout
    char* w = (char*)d_ws;
    float* h0  = (float*)w;  w += (size_t)N * HD * 4;
    float* h1  = (float*)w;  w += (size_t)N * HD * 4;
    float* agg = (float*)w;  w += (size_t)N * HD * 4;
    float* cnt = (float*)w;  w += (((size_t)N * 4) + 255) & ~(size_t)255;
    unsigned short* wbf = (unsigned short*)w;
    unsigned short* Wint = wbf;
    unsigned short* Wl1t = wbf + 32768;
    unsigned short* Wr1t = wbf + 49152;
    unsigned short* Wl2t = wbf + 65536;
    unsigned short* Wr2t = wbf + 81920;

    const int gRow  = (N + 15) / 16;
    const int gZero = (N * HD + 255) / 256;
    const int gEdge = (E + 7) / 8;
    const int gRowW = (N + 7) / 8;

    // 1) weights fp32 -> bf16 (transposed)
    cvt_kernel<<<384, 256, 0, stream>>>(W_in, Wl1, Wr1, Wl2, Wr2, wbf);

    // 2) h0 = x@W_in + b_in + kind_embed + relay*ext_seed
    encode_kernel<<<gRow, 256, 0, stream>>>(x, b_in, node_kind, kind_embed,
                                            ext_seed, Wint, h0, N);

    // 3) mean-aggregate layer 1
    zero_kernel<<<gZero, 256, 0, stream>>>(agg, cnt, N);
    scatter_kernel<<<gEdge, 256, 0, stream>>>(h0, esrc, edst, agg, cnt, E);

    // 4) h1 = (agg/cnt)@Wl1 + h0@Wr1 + bl1
    sage_kernel<<<gRow, 256, 0, stream>>>(agg, cnt, h0, Wl1t, Wr1t, bl1, h1, N);

    // 5) h1 = relu(layernorm(h1)) in place
    ln_relu_kernel<<<gRowW, 256, 0, stream>>>(h1, gamma, beta, N);

    // 6) mean-aggregate layer 2
    zero_kernel<<<gZero, 256, 0, stream>>>(agg, cnt, N);
    scatter_kernel<<<gEdge, 256, 0, stream>>>(h1, esrc, edst, agg, cnt, E);

    // 7) out = (agg/cnt)@Wl2 + h1@Wr2 + bl2
    sage_kernel<<<gRow, 256, 0, stream>>>(agg, cnt, h1, Wl2t, Wr2t, bl2, out, N);
}
